// ZorroPP_26233660244617
// MI455X (gfx1250) — matrix operand order repaired and checked by reading, not compiled, not run on hardware
//
#include <hip/hip_runtime.h>
#include <cmath>

#define BB   512
#define CC   4096
#define KK   1000
#define SKM  8192
#define KPAD 1024

typedef _Float16 v16h __attribute__((ext_vector_type(16)));
typedef _Float16 v8h  __attribute__((ext_vector_type(8)));
typedef _Float16 v4h  __attribute__((ext_vector_type(4)));
typedef float    v8f  __attribute__((ext_vector_type(8)));

// ---------------- Kernel A: xsum[c] = sum_i x[i,c] ----------------
__global__ void zpp_colsum(const float* __restrict__ x, float* __restrict__ xsum) {
    int c = blockIdx.x * blockDim.x + threadIdx.x;
    if (c >= CC) return;
    float s = 0.f;
    for (int i = 0; i < BB; ++i) s += x[(size_t)i * CC + c];
    xsum[c] = s;
}

// ---------------- Kernel B: muS[c] = s * mu_hat[c] ----------------
// mu_hat[c] = ( g*((1-lam)*rowsum(sketch_mu) + 0.01*rowsum(noise)) + g*g*kap*xsum[c]*hcount ) / M
__global__ void zpp_mu(const float* __restrict__ smu, const float* __restrict__ noise,
                       const float* __restrict__ g, const float* __restrict__ xsum,
                       const int* __restrict__ tptr, float hcount, float sconst,
                       float* __restrict__ muS) {
    __shared__ float sh1[256];
    __shared__ float sh2[256];
    const int c   = blockIdx.x;
    const int tid = threadIdx.x;
    const float4* r1 = (const float4*)(smu   + (size_t)c * SKM);
    const float4* r2 = (const float4*)(noise + (size_t)c * SKM);
    float s1 = 0.f, s2 = 0.f;
    #pragma unroll
    for (int j = 0; j < 8; ++j) {              // 8192 floats = 2048 float4 / 256 threads
        float4 a = r1[tid + 256 * j];
        float4 b = r2[tid + 256 * j];
        s1 += a.x + a.y + a.z + a.w;
        s2 += b.x + b.y + b.z + b.w;
    }
    sh1[tid] = s1; sh2[tid] = s2;
    __syncthreads();
    for (int off = 128; off > 0; off >>= 1) {
        if (tid < off) { sh1[tid] += sh1[tid + off]; sh2[tid] += sh2[tid + off]; }
        __syncthreads();
    }
    if (tid == 0) {
        float tf  = (float)(*tptr);
        float lam = 1.f - __expf(-tf / 64.f);
        float kap = 1.f / tf;
        float gc  = g[c];
        float mu  = (gc * ((1.f - lam) * sh1[0] + 0.01f * sh2[0])
                     + gc * gc * kap * xsum[c] * hcount) * (1.f / (float)SKM);
        muS[c] = sconst * mu;
    }
}

// ---------------- Kernel C: cvec[k] = b[k] - sum_c muS[c]*W[c,k] ----------------
__global__ void zpp_cvec(const float* __restrict__ muS, const float* __restrict__ W,
                         const float* __restrict__ b, float* __restrict__ cvec) {
    int k = blockIdx.x * blockDim.x + threadIdx.x;
    if (k >= KK) return;
    float acc = 0.f;
    for (int c = 0; c < CC; ++c) acc += muS[c] * W[(size_t)c * KK + k];
    cvec[k] = b[k] - acc;
}

// ---------------- Kernel D: WhT[k, c] = f16(W[c, k]), zero-padded rows k>=1000 ----
__global__ void zpp_transposeW(const float* __restrict__ W, _Float16* __restrict__ WhT) {
    __shared__ _Float16 tile[32][33];
    const int tx = threadIdx.x;          // 0..31  (k within tile on load)
    const int ty = threadIdx.y;          // 0..7
    const int k0 = blockIdx.x * 32;      // 0..992  (32 tiles cover KPAD=1024)
    const int c0 = blockIdx.y * 32;      // 0..4064
    #pragma unroll
    for (int j = 0; j < 32; j += 8) {
        int c = c0 + ty + j;
        int k = k0 + tx;
        float v = (k < KK) ? W[(size_t)c * KK + k] : 0.f;
        tile[ty + j][tx] = (_Float16)v;  // tile[c_local][k_local]
    }
    __syncthreads();
    #pragma unroll
    for (int j = 0; j < 32; j += 8) {
        int k = k0 + ty + j;
        int c = c0 + tx;
        WhT[(size_t)k * CC + c] = tile[tx][ty + j];
    }
}

// ---------------- Kernel E: xh = f16(s * x), row-major ----------------
__global__ void zpp_xh(const float* __restrict__ x, float sconst, _Float16* __restrict__ xh) {
    int i = blockIdx.x * blockDim.x + threadIdx.x;   // 0 .. (512*4096/4 - 1)
    float4 v = ((const float4*)x)[i];
    v4h o;
    o[0] = (_Float16)(v.x * sconst);
    o[1] = (_Float16)(v.y * sconst);
    o[2] = (_Float16)(v.z * sconst);
    o[3] = (_Float16)(v.w * sconst);
    ((v4h*)xh)[i] = o;
}

// ---------------- Kernel F: out[512,1000] = xh @ WhT^T + cvec (WMMA f16->f32) ----
// Each wave computes a 16x64 output tile: one A fragment reused across 4 WMMAs.
__global__ void zpp_gemm(const _Float16* __restrict__ xh, const _Float16* __restrict__ WhT,
                         const float* __restrict__ cvec, float* __restrict__ out) {
    const int wid  = (blockIdx.x * blockDim.x + threadIdx.x) >> 5;  // 0..511
    const int lane = threadIdx.x & 31;
    const int m0 = (wid >> 4) * 16;     // 32 M-tiles
    const int n0 = (wid & 15) * 64;     // 16 N-quads (covers KPAD=1024)
    const int hs  = lane >> 4;          // half-wave select
    const int l16 = lane & 15;

    // A fragment: M = m0 + l16; K chunks {0..15} (lo half) / {16..31} (hi half)
    const _Float16* arow = xh + (size_t)(m0 + l16) * CC;
    // B fragments: N = n0 + 16*nn + l16; 16 contiguous K per half-wave in WhT rows
    const _Float16* bp0 = WhT + (size_t)(n0 +  0 + l16) * CC + hs * 16;
    const _Float16* bp1 = WhT + (size_t)(n0 + 16 + l16) * CC + hs * 16;
    const _Float16* bp2 = WhT + (size_t)(n0 + 32 + l16) * CC + hs * 16;
    const _Float16* bp3 = WhT + (size_t)(n0 + 48 + l16) * CC + hs * 16;

    v8f acc0 = {}, acc1 = {}, acc2 = {}, acc3 = {};

    for (int k0 = 0; k0 < CC; k0 += 32) {
        __builtin_prefetch(arow + k0 + 64, 0, 1);   // gfx1250 global_prefetch_b8

        union { v16h v; v8h h[2]; } ua, ub0, ub1, ub2, ub3;
        ua.h[0]  = *(const v8h*)(arow + k0 + hs * 16);
        ua.h[1]  = *(const v8h*)(arow + k0 + hs * 16 + 8);
        ub0.h[0] = *(const v8h*)(bp0 + k0);
        ub0.h[1] = *(const v8h*)(bp0 + k0 + 8);
        ub1.h[0] = *(const v8h*)(bp1 + k0);
        ub1.h[1] = *(const v8h*)(bp1 + k0 + 8);
        ub2.h[0] = *(const v8h*)(bp2 + k0);
        ub2.h[1] = *(const v8h*)(bp2 + k0 + 8);
        ub3.h[0] = *(const v8h*)(bp3 + k0);
        ub3.h[1] = *(const v8h*)(bp3 + k0 + 8);

        acc0 = __builtin_amdgcn_wmma_f32_16x16x32_f16(false, ua.v, false, ub0.v, (short)0, acc0, false, false);
        acc1 = __builtin_amdgcn_wmma_f32_16x16x32_f16(false, ua.v, false, ub1.v, (short)0, acc1, false, false);
        acc2 = __builtin_amdgcn_wmma_f32_16x16x32_f16(false, ua.v, false, ub2.v, (short)0, acc2, false, false);
        acc3 = __builtin_amdgcn_wmma_f32_16x16x32_f16(false, ua.v, false, ub3.v, (short)0, acc3, false, false);
    }

    // Epilogue: D VGPR v -> row m0 + v + 8*hs, col n0 + l16 (+16 per tile)
    #pragma unroll
    for (int v = 0; v < 8; ++v) {
        int row = m0 + v + hs * 8;
        float* orow = out + (size_t)row * KK;
        int c0 = n0 + l16;
        if (c0      < KK) orow[c0]      = acc0[v] + cvec[c0];
        if (c0 + 16 < KK) orow[c0 + 16] = acc1[v] + cvec[c0 + 16];
        if (c0 + 32 < KK) orow[c0 + 32] = acc2[v] + cvec[c0 + 32];
        if (c0 + 48 < KK) orow[c0 + 48] = acc3[v] + cvec[c0 + 48];
    }
}

extern "C" void kernel_launch(void* const* d_in, const int* in_sizes, int n_in,
                              void* d_out, int out_size, void* d_ws, size_t ws_size,
                              hipStream_t stream) {
    const float* x     = (const float*)d_in[0];
    const float* smu   = (const float*)d_in[1];
    const float* g     = (const float*)d_in[2];
    const float* noise = (const float*)d_in[3];
    const float* W     = (const float*)d_in[4];
    const float* b     = (const float*)d_in[5];
    // d_in[6] = h: only its element count matters (bincount sums to len(h))
    const int*   tptr  = (const int*)d_in[7];
    const float  hcount = (float)in_sizes[6];
    float* out = (float*)d_out;

    // Chebyshev collapse: xn = s*(x - mu_hat), s independent of t
    const float S = 1.0f / ((1.0f + 1e-6f) * sqrtf(3.0f / 8.0f + 1e-6f));

    // Workspace layout (all 256B-aligned offsets)
    char* ws = (char*)d_ws;
    float*    xsum = (float*)(ws);                       //  16 KB
    float*    muS  = (float*)(ws + 16384);               //  16 KB
    float*    cvec = (float*)(ws + 32768);               //   4 KB
    _Float16* xh   = (_Float16*)(ws + 40960);            //   4 MB  [512 x 4096] f16
    _Float16* WhT  = (_Float16*)(ws + 40960 + 4194304);  //   8 MB  [1024 x 4096] f16

    zpp_colsum    <<<CC / 256, 256, 0, stream>>>(x, xsum);
    zpp_mu        <<<CC, 256, 0, stream>>>(smu, noise, g, xsum, tptr, hcount, S, muS);
    zpp_cvec      <<<4, 256, 0, stream>>>(muS, W, b, cvec);
    zpp_transposeW<<<dim3(KPAD / 32, CC / 32), dim3(32, 8), 0, stream>>>(W, WhT);
    zpp_xh        <<<(BB * CC / 4) / 256, 256, 0, stream>>>(x, S, xh);
    zpp_gemm      <<<64, 256, 0, stream>>>(xh, WhT, cvec, out);   // 512 waves = 32x16 tiles
}
